// MultiHeadSelfAttention_82265803587774
// MI455X (gfx1250) — matrix operand order repaired and checked by reading, not compiled, not run on hardware
//
#include <hip/hip_runtime.h>

#define T_SEQ 2048
#define BATCH 4
#define DIM   1024
#define NH    16
#define DKH   64
#define N3    3072
#define SCALE 0.125f   /* 1/sqrt(64) */

typedef __attribute__((ext_vector_type(16))) __bf16 v16bf;
typedef __attribute__((ext_vector_type(8)))  float  v8f;

union BFrag {
  v16bf v;
  uint4 q[2];
};

__device__ inline unsigned short f2bf_bits(float f) {
  union { __bf16 h; unsigned short u; } p;
  p.h = (__bf16)f;
  return p.u;
}
__device__ inline unsigned int pack2bf(float lo, float hi) {
  union { __bf16 h[2]; unsigned int u; } p;
  p.h[0] = (__bf16)lo;
  p.h[1] = (__bf16)hi;
  return p.u;
}
__device__ inline v8f v8f_zero() {
  v8f z = {0.f, 0.f, 0.f, 0.f, 0.f, 0.f, 0.f, 0.f};
  return z;
}
__device__ inline v8f wmma_bf16(v16bf a, v16bf b, v8f c) {
  return __builtin_amdgcn_wmma_f32_16x16x32_bf16(
      false, a, false, b, (short)0, c, false, false);
}

// --- CDNA5 async LDS copy (ASYNCcnt-tracked DMA into LDS, ISA 15.18 op 98) ---
// lds_addr: wave-relative LDS byte address (low 32 bits of the flat address).
__device__ inline void async_ld_b128(unsigned lds_addr, const void* gaddr) {
  asm volatile("global_load_async_to_lds_b128 %0, %1, off"
               :: "v"(lds_addr), "v"(gaddr)
               : "memory");
}
__device__ inline void wait_async0() {
#if __has_builtin(__builtin_amdgcn_s_wait_asynccnt)
  __builtin_amdgcn_s_wait_asynccnt(0);
#else
  asm volatile("s_wait_asynccnt 0x0" ::: "memory");
#endif
}
__device__ inline unsigned lds_off(const void* p) {
  return (unsigned)(unsigned long long)p;   // flat addr low 32 bits == LDS offset
}

// ---------------------------------------------------------------------------
// Kernel 1: qkv = x @ W_qkv^T  (M=8192, N=3072, K=1024), fp32 in -> bf16 out.
// Scatter as Q[b,h,t,dk], K[b,h,t,dk], V^T[b,h,dk,t].
// Block: 256 threads = 8 waves, tile 128x128, wave tile 32x64, K-step 32.
// ---------------------------------------------------------------------------
__global__ __launch_bounds__(256) void qkv_gemm_kernel(
    const float* __restrict__ x, const float* __restrict__ wq,
    unsigned short* __restrict__ q_ws, unsigned short* __restrict__ k_ws,
    unsigned short* __restrict__ v_ws)
{
  __shared__ unsigned short sA[128][40];
  __shared__ unsigned short sB[128][40];

  const int tid  = threadIdx.x;
  const int lane = tid & 31;
  const int wid  = tid >> 5;
  const int wm   = wid >> 1;
  const int wn   = wid & 1;
  const int half = lane >> 4;
  const int lr   = lane & 15;
  const int m0   = blockIdx.x * 128;
  const int n0   = blockIdx.y * 128;

  v8f acc[2][4];
  for (int i = 0; i < 2; i++)
    for (int j = 0; j < 4; j++) acc[i][j] = v8f_zero();

  const int row = tid >> 1;
  const int seg = (tid & 1) * 16;
  const unsigned abase = (unsigned)(m0 + row) * DIM + seg;
  const unsigned bbase = (unsigned)(n0 + row) * DIM + seg;

  for (int kk = 0; kk < DIM; kk += 32) {
    if (kk + 32 < DIM) {    // L2 prefetch of next K-tile (global_prefetch_b8)
      __builtin_prefetch(x  + abase + kk + 32, 0, 0);
      __builtin_prefetch(wq + bbase + kk + 32, 0, 0);
    }
    {
      const float4* src = (const float4*)(x + abase + kk);
      float4 f0 = src[0], f1 = src[1], f2 = src[2], f3 = src[3];
      uint4 u0, u1;
      u0.x = pack2bf(f0.x, f0.y); u0.y = pack2bf(f0.z, f0.w);
      u0.z = pack2bf(f1.x, f1.y); u0.w = pack2bf(f1.z, f1.w);
      u1.x = pack2bf(f2.x, f2.y); u1.y = pack2bf(f2.z, f2.w);
      u1.z = pack2bf(f3.x, f3.y); u1.w = pack2bf(f3.z, f3.w);
      *(uint4*)&sA[row][seg]     = u0;
      *(uint4*)&sA[row][seg + 8] = u1;
    }
    {
      const float4* src = (const float4*)(wq + bbase + kk);
      float4 f0 = src[0], f1 = src[1], f2 = src[2], f3 = src[3];
      uint4 u0, u1;
      u0.x = pack2bf(f0.x, f0.y); u0.y = pack2bf(f0.z, f0.w);
      u0.z = pack2bf(f1.x, f1.y); u0.w = pack2bf(f1.z, f1.w);
      u1.x = pack2bf(f2.x, f2.y); u1.y = pack2bf(f2.z, f2.w);
      u1.z = pack2bf(f3.x, f3.y); u1.w = pack2bf(f3.z, f3.w);
      *(uint4*)&sB[row][seg]     = u0;
      *(uint4*)&sB[row][seg + 8] = u1;
    }
    __syncthreads();

    BFrag a[2], b[4];
    for (int s = 0; s < 2; s++) {
      const int mr = wm * 32 + s * 16 + lr;
      a[s].q[0] = *(const uint4*)&sA[mr][half * 8];
      a[s].q[1] = *(const uint4*)&sA[mr][half * 8 + 16];
    }
    for (int j = 0; j < 4; j++) {
      const int nr = wn * 64 + j * 16 + lr;
      b[j].q[0] = *(const uint4*)&sB[nr][half * 8];
      b[j].q[1] = *(const uint4*)&sB[nr][half * 8 + 16];
    }
    for (int i = 0; i < 2; i++)
      for (int j = 0; j < 4; j++)
        acc[i][j] = wmma_bf16(a[i].v, b[j].v, acc[i][j]);
    __syncthreads();
  }

  // Epilogue: c/h uniform per 16-wide n-tile -> scalar branch only.
  for (int j = 0; j < 4; j++) {
    const int ngb = n0 + wn * 64 + j * 16;   // uniform
    const int c   = ngb >> 10;               // 0=Q,1=K,2=V
    const int rem = ngb & 1023;
    const int h   = rem >> 6;
    const int dk  = (rem & 63) + lr;
    if (c == 2) {
      for (int i = 0; i < 2; i++)
        for (int r = 0; r < 8; r++) {
          const int mg = m0 + wm * 32 + i * 16 + r + half * 8;
          const int t  = mg >> 2;
          const int bh = (mg & 3) * NH + h;
          v_ws[(unsigned)(bh * DKH + dk) * T_SEQ + t] = f2bf_bits(acc[i][j][r]);
        }
    } else {
      unsigned short* __restrict__ dst = (c == 0) ? q_ws : k_ws;
      for (int i = 0; i < 2; i++)
        for (int r = 0; r < 8; r++) {
          const int mg = m0 + wm * 32 + i * 16 + r + half * 8;
          const int t  = mg >> 2;
          const int bh = (mg & 3) * NH + h;
          dst[(unsigned)(bh * T_SEQ + t) * DKH + dk] = f2bf_bits(acc[i][j][r]);
        }
    }
  }
}

// ---------------------------------------------------------------------------
// Kernel 2: causal flash attention per (b,h, 64-row t-tile).
// 128 threads = 4 waves; double-buffered K/V tiles filled by async DMA.
// ---------------------------------------------------------------------------
__global__ __launch_bounds__(128) void attn_kernel(
    const unsigned short* __restrict__ q_ws, const unsigned short* __restrict__ k_ws,
    const unsigned short* __restrict__ v_ws, unsigned short* __restrict__ o_ws)
{
  __shared__ unsigned short sK[2][64][72];     // [buf][s][d]
  __shared__ unsigned short sV[2][64][72];     // [buf][d][s]
  __shared__ unsigned short sP[4][16][72];     // per-wave P staging

  const int tid  = threadIdx.x;
  const int lane = tid & 31;
  const int w    = tid >> 5;
  const int half = lane >> 4;
  const int lr   = lane & 15;
  const int tt   = blockIdx.x;
  const int bh   = blockIdx.y;
  const int t0   = tt * 64;
  const int tw   = t0 + w * 16;

  BFrag aq[2];
  {
    const unsigned short* qrow = q_ws + (unsigned)(bh * T_SEQ + tw + lr) * DKH;
    for (int f = 0; f < 2; f++) {
      const int kb = f * 32 + half * 8;
      aq[f].q[0] = *(const uint4*)(qrow + kb);
      aq[f].q[1] = *(const uint4*)(qrow + kb + 16);
    }
  }

  v8f oacc[4];
  for (int j = 0; j < 4; j++) oacc[j] = v8f_zero();
  float mrow[8], lsum[8];
  for (int r = 0; r < 8; r++) { mrow[r] = -1e30f; lsum[r] = 0.f; }

  const int crow = tid >> 1;          // 0..63
  const int cseg = (tid & 1) * 32;    // 0|32

  // Issue async DMA fill of K/V tile `sb` into buffer `p`.
  auto issue_kv = [&](int sb, int p) {
    const unsigned kbase = (unsigned)(bh * T_SEQ + sb * 64 + crow) * DKH + cseg;
    const unsigned vbase = (unsigned)(bh * DKH + crow) * T_SEQ + sb * 64 + cseg;
    for (int qd = 0; qd < 4; qd++) {
      async_ld_b128(lds_off(&sK[p][crow][cseg + 8 * qd]), k_ws + kbase + 8 * qd);
      async_ld_b128(lds_off(&sV[p][crow][cseg + 8 * qd]), v_ws + vbase + 8 * qd);
    }
  };

  issue_kv(0, 0);

  for (int sb = 0; sb <= tt; sb++) {
    const int p  = sb & 1;
    const int s0 = sb * 64;
    wait_async0();          // own tile-p fills landed in LDS
    __syncthreads();        // tile p complete block-wide; buffer p^1 retired
    if (sb < tt) issue_kv(sb + 1, p ^ 1);   // DMA next tile during compute

    v8f sc[4];
    for (int nt = 0; nt < 4; nt++) {
      sc[nt] = v8f_zero();
      const int nr = nt * 16 + lr;
      BFrag b0, b1;
      b0.q[0] = *(const uint4*)&sK[p][nr][half * 8];
      b0.q[1] = *(const uint4*)&sK[p][nr][half * 8 + 16];
      b1.q[0] = *(const uint4*)&sK[p][nr][32 + half * 8];
      b1.q[1] = *(const uint4*)&sK[p][nr][32 + half * 8 + 16];
      sc[nt] = wmma_bf16(aq[0].v, b0.v, sc[nt]);
      sc[nt] = wmma_bf16(aq[1].v, b1.v, sc[nt]);
    }

    const bool diag = (sb == tt);
    for (int nt = 0; nt < 4; nt++) {
      const int s_col = s0 + nt * 16 + lr;
      for (int r = 0; r < 8; r++) {
        float v = sc[nt][r] * SCALE;
        if (diag && (s_col > tw + r + half * 8)) v = -1e30f;
        sc[nt][r] = v;
      }
    }

    float alpha[8];
    for (int r = 0; r < 8; r++) {
      float mx = fmaxf(fmaxf(sc[0][r], sc[1][r]), fmaxf(sc[2][r], sc[3][r]));
      mx = fmaxf(mx, __shfl_xor(mx, 1, 32));
      mx = fmaxf(mx, __shfl_xor(mx, 2, 32));
      mx = fmaxf(mx, __shfl_xor(mx, 4, 32));
      mx = fmaxf(mx, __shfl_xor(mx, 8, 32));
      const float mnew = fmaxf(mrow[r], mx);
      alpha[r] = __expf(mrow[r] - mnew);
      mrow[r]  = mnew;
      float rs = 0.f;
      for (int nt = 0; nt < 4; nt++) {
        const float pv = __expf(sc[nt][r] - mnew);
        sc[nt][r] = pv;
        rs += pv;
      }
      rs += __shfl_xor(rs, 1, 32);
      rs += __shfl_xor(rs, 2, 32);
      rs += __shfl_xor(rs, 4, 32);
      rs += __shfl_xor(rs, 8, 32);
      lsum[r] = lsum[r] * alpha[r] + rs;
    }
    for (int j = 0; j < 4; j++)
      for (int r = 0; r < 8; r++)
        oacc[j][r] *= alpha[r];

    for (int nt = 0; nt < 4; nt++)
      for (int r = 0; r < 8; r++)
        sP[w][r + half * 8][nt * 16 + lr] = f2bf_bits(sc[nt][r]);
    __syncthreads();

    for (int f = 0; f < 2; f++) {
      BFrag ap;
      const unsigned short* prow = &sP[w][lr][0];
      const int kb = f * 32 + half * 8;
      ap.q[0] = *(const uint4*)(prow + kb);
      ap.q[1] = *(const uint4*)(prow + kb + 16);
      for (int j = 0; j < 4; j++) {
        BFrag bv;
        const int dr  = j * 16 + lr;
        const int kb2 = f * 32 + half * 8;
        bv.q[0] = *(const uint4*)&sV[p][dr][kb2];
        bv.q[1] = *(const uint4*)&sV[p][dr][kb2 + 16];
        oacc[j] = wmma_bf16(ap.v, bv.v, oacc[j]);
      }
    }
  }

  const int b_idx = bh >> 4;
  const int h     = bh & 15;
  float inv[8];
  for (int r = 0; r < 8; r++) inv[r] = 1.f / lsum[r];
  for (int j = 0; j < 4; j++) {
    const int d = h * 64 + j * 16 + lr;
    for (int r = 0; r < 8; r++) {
      const int t = tw + r + half * 8;
      o_ws[(unsigned)(t * BATCH + b_idx) * DIM + d] = f2bf_bits(oacc[j][r] * inv[r]);
    }
  }
}

// ---------------------------------------------------------------------------
// Kernel 3: out = O @ W_out^T + b_out  (M=8192, N=1024, K=1024), fp32 out.
// A tile (bf16) filled by async DMA, overlapped with B fp32->bf16 convert.
// ---------------------------------------------------------------------------
__global__ __launch_bounds__(256) void out_gemm_kernel(
    const unsigned short* __restrict__ o_ws, const float* __restrict__ wo,
    const float* __restrict__ bias, float* __restrict__ out)
{
  __shared__ unsigned short sA[128][40];
  __shared__ unsigned short sB[128][40];

  const int tid  = threadIdx.x;
  const int lane = tid & 31;
  const int wid  = tid >> 5;
  const int wm   = wid >> 1;
  const int wn   = wid & 1;
  const int half = lane >> 4;
  const int lr   = lane & 15;
  const int m0   = blockIdx.x * 128;
  const int n0   = blockIdx.y * 128;

  v8f acc[2][4];
  for (int i = 0; i < 2; i++)
    for (int j = 0; j < 4; j++) acc[i][j] = v8f_zero();

  const int row = tid >> 1;
  const int seg = (tid & 1) * 16;
  const unsigned abase = (unsigned)(m0 + row) * DIM + seg;
  const unsigned bbase = (unsigned)(n0 + row) * DIM + seg;

  for (int kk = 0; kk < DIM; kk += 32) {
    // A (bf16): async DMA straight into LDS
    async_ld_b128(lds_off(&sA[row][seg]),     o_ws + abase + kk);
    async_ld_b128(lds_off(&sA[row][seg + 8]), o_ws + abase + kk + 8);
    {   // B: fp32 -> bf16 convert (overlaps with async A fill)
      const float4* src = (const float4*)(wo + bbase + kk);
      float4 f0 = src[0], f1 = src[1], f2 = src[2], f3 = src[3];
      uint4 u0, u1;
      u0.x = pack2bf(f0.x, f0.y); u0.y = pack2bf(f0.z, f0.w);
      u0.z = pack2bf(f1.x, f1.y); u0.w = pack2bf(f1.z, f1.w);
      u1.x = pack2bf(f2.x, f2.y); u1.y = pack2bf(f2.z, f2.w);
      u1.z = pack2bf(f3.x, f3.y); u1.w = pack2bf(f3.z, f3.w);
      *(uint4*)&sB[row][seg]     = u0;
      *(uint4*)&sB[row][seg + 8] = u1;
    }
    wait_async0();
    __syncthreads();

    BFrag a[2], b[4];
    for (int s = 0; s < 2; s++) {
      const int mr = wm * 32 + s * 16 + lr;
      a[s].q[0] = *(const uint4*)&sA[mr][half * 8];
      a[s].q[1] = *(const uint4*)&sA[mr][half * 8 + 16];
    }
    for (int j = 0; j < 4; j++) {
      const int nr = wn * 64 + j * 16 + lr;
      b[j].q[0] = *(const uint4*)&sB[nr][half * 8];
      b[j].q[1] = *(const uint4*)&sB[nr][half * 8 + 16];
    }
    for (int i = 0; i < 2; i++)
      for (int j = 0; j < 4; j++)
        acc[i][j] = wmma_bf16(a[i].v, b[j].v, acc[i][j]);
    __syncthreads();
  }

  for (int j = 0; j < 4; j++) {
    const int ng = n0 + wn * 64 + j * 16 + lr;
    const float bv = bias[ng];
    for (int i = 0; i < 2; i++)
      for (int r = 0; r < 8; r++) {
        const int mg = m0 + wm * 32 + i * 16 + r + half * 8;
        out[(unsigned)(mg * DIM + ng)] = acc[i][j][r] + bv;
      }
  }
}

// ---------------------------------------------------------------------------
extern "C" void kernel_launch(void* const* d_in, const int* in_sizes, int n_in,
                              void* d_out, int out_size, void* d_ws, size_t ws_size,
                              hipStream_t stream) {
  (void)in_sizes; (void)n_in; (void)out_size; (void)ws_size;
  const float* x    = (const float*)d_in[0];
  const float* Wqkv = (const float*)d_in[1];
  const float* Wout = (const float*)d_in[2];
  const float* bout = (const float*)d_in[3];
  float* out = (float*)d_out;

  const size_t seg = (size_t)BATCH * NH * T_SEQ * DKH;
  unsigned short* q_ws = (unsigned short*)d_ws;
  unsigned short* k_ws = q_ws + seg;
  unsigned short* v_ws = k_ws + seg;
  unsigned short* o_ws = v_ws + seg;

  qkv_gemm_kernel<<<dim3(64, 24), 256, 0, stream>>>(x, Wqkv, q_ws, k_ws, v_ws);
  attn_kernel<<<dim3(T_SEQ / 64, BATCH * NH), 128, 0, stream>>>(q_ws, k_ws, v_ws, o_ws);
  out_gemm_kernel<<<dim3(64, 8), 256, 0, stream>>>(o_ws, Wout, bout, out);
}
